// TransformerBlock_90692529423016
// MI455X (gfx1250) — matrix operand order repaired and checked by reading, not compiled, not run on hardware
//
#include <hip/hip_runtime.h>
#include <hip/hip_bf16.h>
#include <math.h>

// ---------------------------------------------------------------------------
// Types for CDNA5 WMMA (wave32): v_wmma_f32_16x16x32_bf16
// ---------------------------------------------------------------------------
typedef __bf16 bf16;
typedef __attribute__((ext_vector_type(16))) __bf16 v16bf;
typedef __attribute__((ext_vector_type(8)))  float  v8f;

union FragBF {
    v16bf v;
    uint4 u[2];
};

__device__ __forceinline__ v8f wmma_bf16(const FragBF& a, const FragBF& b, v8f c) {
    // D = A(16x32) * B(32x16) + C(16x16), f32 accumulate
    return __builtin_amdgcn_wmma_f32_16x16x32_bf16(
        /*neg_a=*/false, a.v, /*neg_b=*/false, b.v,
        /*c_mod=*/(short)0, c, /*reuse_a=*/false, /*reuse_b=*/false);
}

__device__ __forceinline__ v8f v8f_zero() {
    v8f z;
#pragma unroll
    for (int i = 0; i < 8; ++i) z[i] = 0.0f;
    return z;
}

// ---------------------------------------------------------------------------
// Problem constants
// ---------------------------------------------------------------------------
#define BB   2
#define SS   4096
#define DD   512
#define HH   8
#define DK   64
#define DFF  2048
#define MSZ  (BB * SS)          // 8192 rows

#define R_KEEP 0.8333333333333333f   // (N_RES-1)/N_RES
#define R_NEW  0.1666666666666667f   // 1/N_RES
#define GELU_INV 0.8858181415537249f // 1/1.1289

// ---------------------------------------------------------------------------
// fp32 -> bf16 conversion
// ---------------------------------------------------------------------------
__global__ void cvt_f32_bf16(const float* __restrict__ src, bf16* __restrict__ dst, int n) {
    int i = blockIdx.x * 256 + threadIdx.x;
    if (i < n) dst[i] = (bf16)src[i];
}

// ---------------------------------------------------------------------------
// Tiled WMMA GEMM:  C[M,N] = A[M,K] (bf16, row-major) @ W[N,K]^T (bf16)
// Block tile 128x128, K-step 32. 8 waves: wave (wm 0..3, wn 0..1) owns 32x64.
// MODE 0: store bf16 C                      (QKV projections)
// MODE 1: xn = 5/6*resid + 1/6*(C/3) -> outf (fp32) and outb (bf16)
// MODE 2: store bf16 gelu_exact(C)/1.1289   (MLP up)
// MODE 3: outf = 5/6*resid + 1/6*C  (fp32)  (MLP down -> d_out)
// ---------------------------------------------------------------------------
#define BK 32
#define LDSA 40   // padded bf16 row stride (80B) -> conflict-free b128 reads

template <int MODE>
__global__ __launch_bounds__(256) void gemm_bf16_wmma(
    const bf16* __restrict__ A, const bf16* __restrict__ W,
    int M, int N, int K,
    const float* __restrict__ resid, float* __restrict__ outf, bf16* __restrict__ outb) {
    (void)M;
    __shared__ bf16 As[128 * LDSA];
    __shared__ bf16 Ws[128 * LDSA];

    const int tid  = threadIdx.x;
    const int wid  = tid >> 5;
    const int lane = tid & 31;
    const int lm   = lane & 15;
    const int lh   = lane >> 4;
    const int wm   = wid >> 1;   // 0..3 -> 32-row slab
    const int wn   = wid & 1;    // 0..1 -> 64-col slab

    const int row0 = blockIdx.y * 128;
    const int col0 = blockIdx.x * 128;

    v8f acc[2][4];
#pragma unroll
    for (int mi = 0; mi < 2; ++mi)
#pragma unroll
        for (int ni = 0; ni < 4; ++ni) acc[mi][ni] = v8f_zero();

    for (int k0 = 0; k0 < K; k0 += BK) {
        // cooperative tile load: 128 rows x 32 bf16 = 512 x 16B chunks each
#pragma unroll
        for (int it = 0; it < 2; ++it) {
            int c  = tid + it * 256;
            int r  = c >> 2;
            int kc = (c & 3) << 3;
            *(uint4*)&As[r * LDSA + kc] =
                *(const uint4*)(A + (size_t)(row0 + r) * K + k0 + kc);
            *(uint4*)&Ws[r * LDSA + kc] =
                *(const uint4*)(W + (size_t)(col0 + r) * K + k0 + kc);
            if (k0 + BK < K) {  // hint next K-tile into cache
                __builtin_prefetch(A + (size_t)(row0 + r) * K + k0 + BK + kc, 0, 1);
                __builtin_prefetch(W + (size_t)(col0 + r) * K + k0 + BK + kc, 0, 1);
            }
        }
        __syncthreads();

        // A fragments: lane holds row (wave row + lm), K chunks lh*8 and lh*8+16
        FragBF af[2];
#pragma unroll
        for (int mi = 0; mi < 2; ++mi) {
            const bf16* p = &As[(wm * 32 + mi * 16 + lm) * LDSA + lh * 8];
            af[mi].u[0] = *(const uint4*)p;
            af[mi].u[1] = *(const uint4*)(p + 16);
        }
        // B fragments: lane holds W row (wave col + lm), K chunks lh*8 and lh*8+16
        FragBF bf[4];
#pragma unroll
        for (int ni = 0; ni < 4; ++ni) {
            const bf16* p = &Ws[(wn * 64 + ni * 16 + lm) * LDSA + lh * 8];
            bf[ni].u[0] = *(const uint4*)p;
            bf[ni].u[1] = *(const uint4*)(p + 16);
        }
#pragma unroll
        for (int mi = 0; mi < 2; ++mi)
#pragma unroll
            for (int ni = 0; ni < 4; ++ni)
                acc[mi][ni] = wmma_bf16(af[mi], bf[ni], acc[mi][ni]);
        __syncthreads();
    }

    // epilogue: C element (M = r + 8*lh, N = lm) per VGPR r
#pragma unroll
    for (int mi = 0; mi < 2; ++mi)
#pragma unroll
        for (int ni = 0; ni < 4; ++ni)
#pragma unroll
            for (int r = 0; r < 8; ++r) {
                int gr = row0 + wm * 32 + mi * 16 + r + (lh << 3);
                int gc = col0 + wn * 64 + ni * 16 + lm;
                size_t idx = (size_t)gr * N + gc;
                float c = acc[mi][ni][r];
                if (MODE == 0) {
                    outb[idx] = (bf16)c;
                } else if (MODE == 1) {
                    float xn = R_KEEP * resid[idx] + R_NEW * (c * (1.0f / 3.0f));
                    outf[idx] = xn;
                    outb[idx] = (bf16)xn;
                } else if (MODE == 2) {
                    float g = 0.5f * c * (1.0f + erff(c * 0.70710678118654752f));
                    outb[idx] = (bf16)(g * GELU_INV);
                } else {
                    outf[idx] = R_KEEP * resid[idx] + R_NEW * c;
                }
            }
}

// ---------------------------------------------------------------------------
// Flash attention (causal), bf16 in/out, fp32 online softmax.
// grid = (S/128, B*H); block = 256 (8 waves x 16 query rows).
// Key blocks of 64 staged in LDS; V stored transposed; P re-laid out via LDS.
// ---------------------------------------------------------------------------
#define KSTR 72  // padded LDS row stride (144B)

__global__ __launch_bounds__(256) void attn_wmma(
    const bf16* __restrict__ q, const bf16* __restrict__ k,
    const bf16* __restrict__ v, bf16* __restrict__ o) {
    __shared__ bf16 Ks[64 * KSTR];           // [key][dk]
    __shared__ bf16 Vs[64 * KSTR];           // [dk][key] (transposed)
    __shared__ bf16 Ps[8 * 16 * KSTR];       // wave-private P tiles [16][64]

    const int tid  = threadIdx.x;
    const int wid  = tid >> 5;
    const int lane = tid & 31;
    const int lm   = lane & 15;
    const int lh   = lane >> 4;

    const int bh = blockIdx.y;               // b*H + h
    const int b  = bh >> 3;
    const int h  = bh & 7;
    const int q0 = blockIdx.x * 128;
    const int wq0 = q0 + wid * 16;

    const size_t base = (size_t)b * SS * DD + (size_t)h * DK;
    const bf16* qp = q + base;
    const bf16* kp = k + base;
    const bf16* vp = v + base;

    // Q fragments (16 rows x 64 dk -> two K=32 fragments), loaded once
    FragBF qf[2];
#pragma unroll
    for (int kk = 0; kk < 2; ++kk) {
        const bf16* p = qp + (size_t)(wq0 + lm) * DD + kk * 32 + lh * 8;
        qf[kk].u[0] = *(const uint4*)p;
        qf[kk].u[1] = *(const uint4*)(p + 16);
    }

    v8f o_acc[4];
#pragma unroll
    for (int ni = 0; ni < 4; ++ni) o_acc[ni] = v8f_zero();
    float m_i[8], l_i[8];
#pragma unroll
    for (int r = 0; r < 8; ++r) { m_i[r] = -3.0e38f; l_i[r] = 0.0f; }

    const int nkb = (q0 + 128) >> 6;         // causal: keys up to q0+127
    for (int jb = 0; jb < nkb; ++jb) {
        const int j0 = jb << 6;
        __syncthreads();
        // stage K block [64 keys][64 dk] and V block transposed [64 dk][64 keys]
#pragma unroll
        for (int it = 0; it < 2; ++it) {
            int c   = tid + it * 256;
            int r   = c >> 3;
            int dk0 = (c & 7) << 3;
            *(uint4*)&Ks[r * KSTR + dk0] =
                *(const uint4*)(kp + (size_t)(j0 + r) * DD + dk0);
            uint4 d = *(const uint4*)(vp + (size_t)(j0 + r) * DD + dk0);
            alignas(16) bf16 tmp[8];
            *(uint4*)tmp = d;
#pragma unroll
            for (int e = 0; e < 8; ++e) Vs[(dk0 + e) * KSTR + r] = tmp[e];
        }
        __syncthreads();

        // scores S = Q @ K^T  (16 x 64 per wave)
        v8f sc[4];
#pragma unroll
        for (int ni = 0; ni < 4; ++ni) {
            v8f a = v8f_zero();
#pragma unroll
            for (int kk = 0; kk < 2; ++kk) {
                FragBF kf;
                const bf16* p = &Ks[(ni * 16 + lm) * KSTR + kk * 32 + lh * 8];
                kf.u[0] = *(const uint4*)p;
                kf.u[1] = *(const uint4*)(p + 16);
                a = wmma_bf16(qf[kk], kf, a);
            }
            sc[ni] = a;
        }

        // scale + causal mask
#pragma unroll
        for (int ni = 0; ni < 4; ++ni) {
            int kg = j0 + ni * 16 + lm;
#pragma unroll
            for (int r = 0; r < 8; ++r) {
                int qg = wq0 + r + (lh << 3);
                float s = sc[ni][r] * (1.0f / (float)DK);
                sc[ni][r] = (kg > qg) ? -1.0e9f : s;
            }
        }

        // online softmax (rows live in 16-lane halves of the wave)
        float alpha[8];
#pragma unroll
        for (int r = 0; r < 8; ++r) {
            float mx = sc[0][r];
#pragma unroll
            for (int ni = 1; ni < 4; ++ni) mx = fmaxf(mx, sc[ni][r]);
#pragma unroll
            for (int m = 1; m < 16; m <<= 1) mx = fmaxf(mx, __shfl_xor(mx, m, 32));
            float mnew = fmaxf(m_i[r], mx);
            float a = __expf(m_i[r] - mnew);
            float sum = 0.0f;
#pragma unroll
            for (int ni = 0; ni < 4; ++ni) {
                float p = __expf(sc[ni][r] - mnew);
                sc[ni][r] = p;
                sum += p;
            }
#pragma unroll
            for (int m = 1; m < 16; m <<= 1) sum += __shfl_xor(sum, m, 32);
            l_i[r] = l_i[r] * a + sum;
            m_i[r] = mnew;
            alpha[r] = a;
        }
#pragma unroll
        for (int ni = 0; ni < 4; ++ni)
#pragma unroll
            for (int r = 0; r < 8; ++r) o_acc[ni][r] *= alpha[r];

        // re-layout P (C-layout -> A-layout) through wave-private LDS
        bf16* pw = &Ps[wid * 16 * KSTR];
#pragma unroll
        for (int ni = 0; ni < 4; ++ni)
#pragma unroll
            for (int r = 0; r < 8; ++r)
                pw[(r + (lh << 3)) * KSTR + ni * 16 + lm] = (bf16)sc[ni][r];

        // O += P @ V
#pragma unroll
        for (int kk = 0; kk < 2; ++kk) {
            FragBF pf;
            const bf16* pp = &pw[lm * KSTR + kk * 32 + lh * 8];
            pf.u[0] = *(const uint4*)pp;
            pf.u[1] = *(const uint4*)(pp + 16);
#pragma unroll
            for (int ni = 0; ni < 4; ++ni) {
                FragBF vf;
                const bf16* vpp = &Vs[(ni * 16 + lm) * KSTR + kk * 32 + lh * 8];
                vf.u[0] = *(const uint4*)vpp;
                vf.u[1] = *(const uint4*)(vpp + 16);
                o_acc[ni] = wmma_bf16(pf, vf, o_acc[ni]);
            }
        }
    }

    // normalize + store bf16 output (per-head columns of [B*S, D])
#pragma unroll
    for (int ni = 0; ni < 4; ++ni)
#pragma unroll
        for (int r = 0; r < 8; ++r) {
            int qg = wq0 + r + (lh << 3);
            float val = o_acc[ni][r] / l_i[r];
            o[(size_t)(b * SS + qg) * DD + h * DK + ni * 16 + lm] = (bf16)val;
        }
}

// ---------------------------------------------------------------------------
// Host: pipeline launcher
// ---------------------------------------------------------------------------
extern "C" void kernel_launch(void* const* d_in, const int* in_sizes, int n_in,
                              void* d_out, int out_size, void* d_ws, size_t ws_size,
                              hipStream_t stream) {
    (void)in_sizes; (void)n_in; (void)out_size; (void)ws_size;
    const float* x     = (const float*)d_in[0];
    /* d_in[1] = causal mask (int) -- handled analytically */
    const float* wq    = (const float*)d_in[2];
    const float* wk    = (const float*)d_in[3];
    const float* wv    = (const float*)d_in[4];
    const float* wo    = (const float*)d_in[5];
    const float* wup   = (const float*)d_in[6];
    const float* wdown = (const float*)d_in[7];
    float* out = (float*)d_out;

    char* p = (char*)d_ws;
    auto alloc = [&](size_t bytes) -> char* {
        char* r = p;
        p += (bytes + 255) & ~(size_t)255;
        return r;
    };
    const size_t nX  = (size_t)MSZ * DD;    // 8192*512
    const size_t nW  = (size_t)DD * DD;     // 512*512
    const size_t nWU = (size_t)DFF * DD;    // 2048*512
    const size_t nH  = (size_t)MSZ * DFF;   // 8192*2048

    bf16* xb     = (bf16*)alloc(nX * 2);
    bf16* wqb    = (bf16*)alloc(nW * 2);
    bf16* wkb    = (bf16*)alloc(nW * 2);
    bf16* wvb    = (bf16*)alloc(nW * 2);
    bf16* wob    = (bf16*)alloc(nW * 2);
    bf16* wupb   = (bf16*)alloc(nWU * 2);
    bf16* wdownb = (bf16*)alloc(nWU * 2);
    bf16* qb     = (bf16*)alloc(nX * 2);
    bf16* kb     = (bf16*)alloc(nX * 2);
    bf16* vb     = (bf16*)alloc(nX * 2);
    bf16* ab     = (bf16*)alloc(nX * 2);    // attention output (pre-Wo)
    bf16* x1b    = (bf16*)alloc(nX * 2);    // post-attn residual, bf16
    float* x1f   = (float*)alloc(nX * 4);   // post-attn residual, fp32
    bf16* hb     = (bf16*)alloc(nH * 2);    // MLP hidden

    auto cvt = [&](const float* s, bf16* d, size_t n) {
        cvt_f32_bf16<<<dim3((unsigned)((n + 255) / 256)), dim3(256), 0, stream>>>(s, d, (int)n);
    };
    cvt(x, xb, nX);
    cvt(wq, wqb, nW);
    cvt(wk, wkb, nW);
    cvt(wv, wvb, nW);
    cvt(wo, wob, nW);
    cvt(wup, wupb, nWU);
    cvt(wdown, wdownb, nWU);

    const dim3 blk(256);
    // QKV projections: [8192,512] = x @ Wq^T etc.
    gemm_bf16_wmma<0><<<dim3(DD / 128, MSZ / 128), blk, 0, stream>>>(
        xb, wqb, MSZ, DD, DD, nullptr, nullptr, qb);
    gemm_bf16_wmma<0><<<dim3(DD / 128, MSZ / 128), blk, 0, stream>>>(
        xb, wkb, MSZ, DD, DD, nullptr, nullptr, kb);
    gemm_bf16_wmma<0><<<dim3(DD / 128, MSZ / 128), blk, 0, stream>>>(
        xb, wvb, MSZ, DD, DD, nullptr, nullptr, vb);

    // causal flash attention
    attn_wmma<<<dim3(SS / 128, BB * HH), blk, 0, stream>>>(qb, kb, vb, ab);

    // output projection + residual: x1 = 5/6 x + 1/6 * (1/3 * ab @ Wo^T)
    gemm_bf16_wmma<1><<<dim3(DD / 128, MSZ / 128), blk, 0, stream>>>(
        ab, wob, MSZ, DD, DD, x, x1f, x1b);

    // MLP up + exact GELU / 1.1289
    gemm_bf16_wmma<2><<<dim3(DFF / 128, MSZ / 128), blk, 0, stream>>>(
        x1b, wupb, MSZ, DFF, DD, nullptr, nullptr, hb);

    // MLP down + residual -> fp32 output
    gemm_bf16_wmma<3><<<dim3(DD / 128, MSZ / 128), blk, 0, stream>>>(
        hb, wdownb, MSZ, DD, DFF, x1f, out, nullptr);
}
